// RWKVBlock_v3_85066122264889
// MI455X (gfx1250) — matrix operand order repaired and checked by reading, not compiled, not run on hardware
//
#include <hip/hip_runtime.h>

// ---------------------------------------------------------------------------
// RWKV vision block (VRWKV / v6) for gfx1250 — WMMA f16 GEMMs with async-LDS
// B-panel staging + 3-pass segmented WKV scan.
// ---------------------------------------------------------------------------

typedef __attribute__((ext_vector_type(16))) _Float16 v16h;
typedef __attribute__((ext_vector_type(8)))  float    v8f;

#define B_   8
#define C_   128
#define H_   64
#define W_   64
#define T_   4096          // H*W
#define CK_  256
#define M_   32768         // B*T
#define NH_  8
#define HS_  32
#define EPS_ 1e-5f

// ---------------------------------------------------------------------------
// qshift: x viewed as (B,T,CK), t reinterpreted as (h=t/64, w=t%64), quarter
// channels shift left/right/up/down with zero padding.
// ---------------------------------------------------------------------------
__device__ __forceinline__ float qshift_delta(const float* __restrict__ X,
                                              long row, int ck, float center) {
  int t = (int)(row & (T_ - 1));
  int q = ck >> 6;           // CK/4 = 64 channels per quarter
  int nt; bool ok;
  if (q == 0)      { ok = (t & 63) != 0;      nt = t - 1;  }
  else if (q == 1) { ok = (t & 63) != 63;     nt = t + 1;  }
  else if (q == 2) { ok = t >= 64;            nt = t - 64; }
  else             { ok = t < (T_ - 64);      nt = t + 64; }
  float nv = ok ? X[(row - t + nt) * (long)CK_ + ck] : 0.f;
  return nv - center;
}

// ---------------------------------------------------------------------------
// WMMA fragment loaders (ISA 7.12.2 16-bit layouts, wave32)
// ---------------------------------------------------------------------------
__device__ __forceinline__ v16h ld_a_frag(const _Float16* __restrict__ A, int lda,
                                          long m0, int k0, int lane) {
  int r = lane & 15, kh = lane >> 4;
  const _Float16* p = A + (m0 + r) * (long)lda + k0 + kh * 8;
  union { v16h v; float4 q[2]; } u;
  u.q[0] = *(const float4*)(p);
  u.q[1] = *(const float4*)(p + 16);
  return u.v;
}
__device__ __forceinline__ v16h ld_b_lds(const _Float16* Bs, int K,
                                         int nloc, int k0, int lane) {
  int n = lane & 15, kh = lane >> 4;
  const _Float16* p = Bs + (nloc + n) * K + k0 + kh * 8;
  union { v16h v; float4 q[2]; } u;
  u.q[0] = *(const float4*)(p);
  u.q[1] = *(const float4*)(p + 16);
  return u.v;
}

#define WMMA_F16(a, b, c) __builtin_amdgcn_wmma_f32_16x16x32_f16( \
    false, (a), false, (b), (short)0, (c), false, false)

// ---------------------------------------------------------------------------
// Generic WMMA GEMM: C[M,N] = A[M,K] @ B[K,N]. A f16 row-major (lda), B f16
// pre-transposed N-major (ldb == K). Block = 256 threads = 8 waves, block tile
// 256(M) x 64(N); wave tile 32(M) x 64(N) (8 accumulators). The 64xK B panel
// is staged to LDS once per block via gfx1250 async global->LDS copies
// (ASYNCcnt), then B fragments come from LDS in groups of four (8 ds_loads ->
// 8 back-to-back WMMAs) while A fragments alternate between two register sets
// (explicit 2-stage software pipeline, no copy chain). grid = (N/64, M/256).
// Fused epilogues:
//  0: f32                      1: silu -> f32         2: tanh -> f16
//  3: lora-mix -> f16 (aux0 = X32 base, aux1 = coef)  4: exp(-exp(aux1+v)) f32
//  5: f16                      6: relu^2 -> f16       7: sigmoid(v)*aux0 f32
// ---------------------------------------------------------------------------
__global__ __launch_bounds__(256)
void gemm_wmma(const _Float16* __restrict__ A, int lda,
               const _Float16* __restrict__ Bt,
               void* __restrict__ C, int ldc, int K, int mode,
               const void* __restrict__ aux0, const float* __restrict__ aux1) {
  extern __shared__ char smem[];                 // 64 * K halfs
  _Float16* Bs = (_Float16*)smem;

  int lane = threadIdx.x & 31;
  int wid  = threadIdx.x >> 5;
  long m0  = (long)blockIdx.y * 256 + wid * 32;
  int  n0  = blockIdx.x * 64;

  // ---- stage B panel (rows n0..n0+63, all K) into LDS asynchronously ------
  {
    const char* gsrc = (const char*)(Bt + (long)n0 * K);
    int bytes = 128 * K;                         // 64 rows * K * 2B
    for (int off = threadIdx.x * 16; off < bytes; off += 256 * 16) {
      unsigned lds_addr = (unsigned)(size_t)(smem + off);   // low 32b = LDS off
      const char* ga = gsrc + off;
      asm volatile("global_load_async_to_lds_b128 %0, %1, off"
                   :: "v"(lds_addr), "v"(ga) : "memory");
    }
    asm volatile("s_wait_asynccnt 0" ::: "memory");
    __syncthreads();
  }

  v8f acc[8] = {};
  v16h a0 = ld_a_frag(A, lda, m0,      0, lane);
  v16h a1 = ld_a_frag(A, lda, m0 + 16, 0, lane);
  for (int k0 = 0; k0 < K; k0 += 64) {
    // stage 1: B frags for k0, prefetch A frags for k0+32 into c0/c1
    v16h b0 = ld_b_lds(Bs, K,  0, k0, lane);
    v16h b1 = ld_b_lds(Bs, K, 16, k0, lane);
    v16h b2 = ld_b_lds(Bs, K, 32, k0, lane);
    v16h b3 = ld_b_lds(Bs, K, 48, k0, lane);
    v16h c0, c1;
    bool more = (k0 + 32) < K;
    if (more) {
      __builtin_prefetch(A + m0 * (long)lda + k0 + 64, 0, 1);
      c0 = ld_a_frag(A, lda, m0,      k0 + 32, lane);
      c1 = ld_a_frag(A, lda, m0 + 16, k0 + 32, lane);
    }
    acc[0] = WMMA_F16(a0, b0, acc[0]);  acc[4] = WMMA_F16(a1, b0, acc[4]);
    acc[1] = WMMA_F16(a0, b1, acc[1]);  acc[5] = WMMA_F16(a1, b1, acc[5]);
    acc[2] = WMMA_F16(a0, b2, acc[2]);  acc[6] = WMMA_F16(a1, b2, acc[6]);
    acc[3] = WMMA_F16(a0, b3, acc[3]);  acc[7] = WMMA_F16(a1, b3, acc[7]);
    if (more) {
      // stage 2: B frags for k0+32, prefetch A frags for k0+64 back into a0/a1
      b0 = ld_b_lds(Bs, K,  0, k0 + 32, lane);
      b1 = ld_b_lds(Bs, K, 16, k0 + 32, lane);
      b2 = ld_b_lds(Bs, K, 32, k0 + 32, lane);
      b3 = ld_b_lds(Bs, K, 48, k0 + 32, lane);
      if (k0 + 64 < K) {
        a0 = ld_a_frag(A, lda, m0,      k0 + 64, lane);
        a1 = ld_a_frag(A, lda, m0 + 16, k0 + 64, lane);
      }
      acc[0] = WMMA_F16(c0, b0, acc[0]);  acc[4] = WMMA_F16(c1, b0, acc[4]);
      acc[1] = WMMA_F16(c0, b1, acc[1]);  acc[5] = WMMA_F16(c1, b1, acc[5]);
      acc[2] = WMMA_F16(c0, b2, acc[2]);  acc[6] = WMMA_F16(c1, b2, acc[6]);
      acc[3] = WMMA_F16(c0, b3, acc[3]);  acc[7] = WMMA_F16(c1, b3, acc[7]);
    }
  }

  int cr = lane & 15, hi = lane >> 4;
#pragma unroll
  for (int half = 0; half < 2; ++half) {
#pragma unroll
    for (int j = 0; j < 4; ++j) {
      v8f a = acc[half * 4 + j];
#pragma unroll
      for (int i = 0; i < 8; ++i) {
        long row = m0 + half * 16 + hi * 8 + i;
        int  col = n0 + j * 16 + cr;
        long off = row * (long)ldc + col;
        float v  = a[i];
        switch (mode) {
          case 0: ((float*)C)[off] = v; break;
          case 1: ((float*)C)[off] = v / (1.f + expf(-v)); break;
          case 2: ((_Float16*)C)[off] = (_Float16)tanhf(v); break;
          case 3: {
            const float* X = (const float*)aux0;
            float xv = X[row * (long)CK_ + col];
            float d  = qshift_delta(X, row, col, xv);
            ((_Float16*)C)[off] = (_Float16)(xv + d * (aux1[col] + v));
          } break;
          case 4: ((float*)C)[off] = expf(-expf(aux1[col] + v)); break;
          case 5: ((_Float16*)C)[off] = (_Float16)v; break;
          case 6: { float r = fmaxf(v, 0.f);
                    ((_Float16*)C)[off] = (_Float16)(r * r); } break;
          case 7: ((float*)C)[off] =
                    (1.f / (1.f + expf(-v))) * ((const float*)aux0)[off]; break;
        }
      }
    }
  }
}

// ---------------------------------------------------------------------------
// Weight prep: (K,N) f32 row-major -> (Npad,K) f16 N-major, zero-padded rows.
// ---------------------------------------------------------------------------
__global__ void wprep(const float* __restrict__ src, _Float16* __restrict__ dst,
                      int K, int N, int Npad) {
  long idx = (long)blockIdx.x * 256 + threadIdx.x;
  if (idx >= (long)Npad * K) return;
  int n = (int)(idx / K), k = (int)(idx % K);
  dst[idx] = (n < N) ? (_Float16)src[(long)k * N + n] : (_Float16)0.f;
}

// ---------------------------------------------------------------------------
// GroupNorm(1,C) stats per batch: mean + rsqrt(var+eps)
// ---------------------------------------------------------------------------
__global__ void gn_stats(const float* __restrict__ x, long n_per,
                         float* __restrict__ out) {
  int b = blockIdx.x, tid = threadIdx.x;
  const float* p = x + (long)b * n_per;
  float s = 0.f, s2 = 0.f;
  for (long i = tid; i < n_per; i += 256) { float v = p[i]; s += v; s2 += v * v; }
  __shared__ float sh0[256], sh1[256];
  sh0[tid] = s; sh1[tid] = s2; __syncthreads();
  for (int st = 128; st; st >>= 1) {
    if (tid < st) { sh0[tid] += sh0[tid + st]; sh1[tid] += sh1[tid + st]; }
    __syncthreads();
  }
  if (!tid) {
    float mu = sh0[0] / (float)n_per;
    float var = sh1[0] / (float)n_per - mu * mu;
    out[b * 2] = mu; out[b * 2 + 1] = rsqrtf(var + EPS_);
  }
}

// xn = gn(x)*w+b, plus per-(b,c) spatial max (SE pool). grid = B*C, 256 thr
__global__ void xnorm_semax(const float* __restrict__ x,
                            const float* __restrict__ stats,
                            const float* __restrict__ lw,
                            const float* __restrict__ lb,
                            float* __restrict__ xn, float* __restrict__ xmax) {
  int bc = blockIdx.x, tid = threadIdx.x;
  int b = bc >> 7, c = bc & 127;
  float mu = stats[b * 2], rs = stats[b * 2 + 1];
  float w = lw[c], bb = lb[c];
  const float* p = x + (long)bc * T_;
  float* q = xn + (long)bc * T_;
  float mx = -1e30f;
  for (int i = tid; i < T_; i += 256) {
    float v = (p[i] - mu) * rs * w + bb;
    q[i] = v; mx = fmaxf(mx, v);
  }
  __shared__ float sh[256];
  sh[tid] = mx; __syncthreads();
  for (int st = 128; st; st >>= 1) {
    if (tid < st) sh[tid] = fmaxf(sh[tid], sh[tid + st]);
    __syncthreads();
  }
  if (!tid) xmax[bc] = sh[0];
}

// SE MLP: gate[b,c] = sigmoid(W2 relu(W1 mx + b1) + b2). grid = B, 128 thr
__global__ void se_mlp(const float* __restrict__ xmax,
                       const float* __restrict__ w1, const float* __restrict__ b1,
                       const float* __restrict__ w2, const float* __restrict__ b2,
                       float* __restrict__ gate) {
  int b = blockIdx.x, tid = threadIdx.x;
  __shared__ float mx[128], hid[32];
  mx[tid] = xmax[b * 128 + tid]; __syncthreads();
  if (tid < 32) {
    float a = b1[tid];
    for (int i = 0; i < 128; ++i) a += w1[tid * 128 + i] * mx[i];
    hid[tid] = fmaxf(a, 0.f);
  }
  __syncthreads();
  float a = b2[tid];
  for (int j = 0; j < 32; ++j) a += w2[tid * 32 + j] * hid[j];
  gate[b * 128 + tid] = 1.f / (1.f + expf(-a));
}

// cnd0[b,c,hw] = cond1_w[c]*cond[b,hw] + cond1_b[c]
__global__ void cond1x1(const float* __restrict__ cond,
                        const float* __restrict__ w, const float* __restrict__ bb,
                        float* __restrict__ cnd0) {
  long idx = (long)blockIdx.x * 256 + threadIdx.x;
  if (idx >= (long)B_ * C_ * T_) return;
  int hw = (int)(idx & (T_ - 1));
  long bc = idx >> 12; int c = (int)(bc & 127); int b = (int)(bc >> 7);
  cnd0[idx] = w[c] * cond[(long)b * T_ + hw] + bb[c];
}

// s = (dw3x3(xn; xconv) + dw3x3(gn(cnd0); conddw)) * silu(gate * xn)
__global__ void fuse_spatial(const float* __restrict__ xn,
                             const float* __restrict__ cnd0,
                             const float* __restrict__ statsC,
                             const float* __restrict__ xconv,
                             const float* __restrict__ condw,
                             const float* __restrict__ clnw,
                             const float* __restrict__ clnb,
                             const float* __restrict__ gate,
                             float* __restrict__ sout) {
  long idx = (long)blockIdx.x * 256 + threadIdx.x;
  if (idx >= (long)B_ * C_ * T_) return;
  int wq = (int)(idx & 63), h = (int)((idx >> 6) & 63);
  long bc = idx >> 12; int c = (int)(bc & 127); int b = (int)(bc >> 7);
  float cmu = statsC[b * 2], crs = statsC[b * 2 + 1];
  float gw = clnw[c] * crs;
  float gb = clnb[c] - cmu * gw;
  const float* xp = xn + bc * T_;
  const float* cp = cnd0 + bc * T_;
  float accx = 0.f, accc = 0.f;
#pragma unroll
  for (int dy = 0; dy < 3; ++dy) {
    int hh = h + dy - 1;
    if (hh < 0 || hh > 63) continue;
#pragma unroll
    for (int dx = 0; dx < 3; ++dx) {
      int ww = wq + dx - 1;
      if (ww < 0 || ww > 63) continue;
      int off = hh * 64 + ww;
      accx += xp[off] * xconv[c * 9 + dy * 3 + dx];
      accc += (cp[off] * gw + gb) * condw[c * 9 + dy * 3 + dx];
    }
  }
  float xc = xp[h * 64 + wq];
  float pre = gate[bc] * xc;
  float xact = pre / (1.f + expf(-pre));
  sout[idx] = (accx + accc) * xact;
}

// cross-scan K=2, SELECT=1 (column-major + reverse). X32 (B,T,CK) token-major.
__global__ void cross_scan(const float* __restrict__ s, float* __restrict__ X) {
  long idx = (long)blockIdx.x * 256 + threadIdx.x;
  if (idx >= (long)B_ * T_ * CK_) return;
  int ck = (int)(idx & 255);
  int t  = (int)((idx >> 8) & (T_ - 1));
  int b  = (int)(idx >> 20);
  int c  = ck & 127;
  int tt = (ck < 128) ? t : (T_ - 1 - t);
  X[idx] = s[((long)(b * C_ + c)) * T_ + (tt & 63) * 64 + (tt >> 6)];
}

// out16 = f16(src + (qshift(src)-src)*coef[ck])
__global__ void mix16(const float* __restrict__ src, const float* __restrict__ coef,
                      _Float16* __restrict__ dst) {
  long idx = (long)blockIdx.x * 256 + threadIdx.x;
  if (idx >= (long)M_ * CK_) return;
  int ck = (int)(idx & 255);
  long row = idx >> 8;
  float x = src[idx];
  float d = qshift_delta(src, row, ck, x);
  dst[idx] = (_Float16)(x + d * coef[ck]);
}

// ---------------------------------------------------------------------------
// WKV scan, 3-pass segmented (linear diagonal-decay recurrence):
//   S_t = diag(w_t) S_{t-1} + k_t v_t^T ,  y_t = r_t^T (S_{t-1} + u (.) k v^T)
// 64 segments of 64 steps per (b,head). Lane m holds state column S[:,m].
// ---------------------------------------------------------------------------
#define NSEG 64
#define LSEG 64

__global__ __launch_bounds__(32)
void wkv_pass1(const float* __restrict__ k, const float* __restrict__ v,
               const float* __restrict__ w,
               float* __restrict__ Sseg, float* __restrict__ Dseg) {
  int seg = blockIdx.x, bh = blockIdx.y;
  int b = bh >> 3, hd = bh & 7;
  int lane = threadIdx.x;
  float S[32];
#pragma unroll
  for (int n = 0; n < 32; ++n) S[n] = 0.f;
  float D = 1.f;
  long base = ((long)b * T_ + (long)seg * LSEG) * CK_ + hd * HS_;
  for (int tl = 0; tl < LSEG; ++tl) {
    long off = base + (long)tl * CK_;
    float kl = k[off + lane], vl = v[off + lane], wl = w[off + lane];
    D *= wl;
#pragma unroll
    for (int n = 0; n < 32; ++n) {
      float kn = __shfl(kl, n, 32);
      float wn = __shfl(wl, n, 32);
      S[n] = wn * S[n] + kn * vl;
    }
  }
  long sb = ((long)bh * NSEG + seg) * 1024;
#pragma unroll
  for (int n = 0; n < 32; ++n) Sseg[sb + n * 32 + lane] = S[n];
  Dseg[((long)bh * NSEG + seg) * 32 + lane] = D;
}

__global__ __launch_bounds__(32)
void wkv_pass2(const float* __restrict__ Sseg, const float* __restrict__ Dseg,
               float* __restrict__ Sini) {
  int bh = blockIdx.x, lane = threadIdx.x;
  float S[32];
#pragma unroll
  for (int n = 0; n < 32; ++n) S[n] = 0.f;
  for (int seg = 0; seg < NSEG; ++seg) {
    long sb = (long)bh * NSEG + seg;
#pragma unroll
    for (int n = 0; n < 32; ++n) Sini[sb * 1024 + n * 32 + lane] = S[n];
    float Dl = Dseg[sb * 32 + lane];
#pragma unroll
    for (int n = 0; n < 32; ++n) {
      float Dn = __shfl(Dl, n, 32);
      S[n] = Dn * S[n] + Sseg[sb * 1024 + n * 32 + lane];
    }
  }
}

__global__ __launch_bounds__(32)
void wkv_pass3(const float* __restrict__ r, const float* __restrict__ k,
               const float* __restrict__ v, const float* __restrict__ w,
               const float* __restrict__ u, const float* __restrict__ Sini,
               float* __restrict__ y) {
  int seg = blockIdx.x, bh = blockIdx.y;
  int b = bh >> 3, hd = bh & 7;
  int lane = threadIdx.x;
  long sb = ((long)bh * NSEG + seg) * 1024;
  float S[32];
#pragma unroll
  for (int n = 0; n < 32; ++n) S[n] = Sini[sb + n * 32 + lane];
  float ul = u[hd * HS_ + lane];
  float un[32];
#pragma unroll
  for (int n = 0; n < 32; ++n) un[n] = __shfl(ul, n, 32);
  long base = ((long)b * T_ + (long)seg * LSEG) * CK_ + hd * HS_;
  for (int tl = 0; tl < LSEG; ++tl) {
    long off = base + (long)tl * CK_;
    float rl = r[off + lane], kl = k[off + lane];
    float vl = v[off + lane], wl = w[off + lane];
    float acc = 0.f;
#pragma unroll
    for (int n = 0; n < 32; ++n) {
      float rn = __shfl(rl, n, 32);
      float kn = __shfl(kl, n, 32);
      float wn = __shfl(wl, n, 32);
      float kv = kn * vl;
      acc += rn * (S[n] + un[n] * kv);
      S[n] = wn * S[n] + kv;
    }
    y[off + lane] = acc;
  }
}

// per-(token,head) LN over HS=32, then * g, -> f16. grid = M, 256 thr (8 waves)
__global__ void lnx_gate(const float* __restrict__ y, const float* __restrict__ g,
                         const float* __restrict__ lnw, const float* __restrict__ lnb,
                         _Float16* __restrict__ out) {
  long tok = blockIdx.x;
  int tid = threadIdx.x;
  int ck = tid;
  float val = y[tok * CK_ + ck];
  float s = val, s2 = val * val;
  for (int o = 16; o; o >>= 1) {
    s  += __shfl_xor(s,  o, 32);
    s2 += __shfl_xor(s2, o, 32);
  }
  float mu = s * (1.f / 32.f);
  float var = s2 * (1.f / 32.f) - mu * mu;
  float nv = (val - mu) * rsqrtf(var + EPS_) * lnw[ck] + lnb[ck];
  out[tok * CK_ + ck] = (_Float16)(nv * g[tok * CK_ + ck]);
}

// cross-merge + channel LayerNorm (over C) + residual. grid = B*T pixels, 128 thr
__global__ void merge_outln(const float* __restrict__ xf, const float* __restrict__ xin,
                            const float* __restrict__ ow, const float* __restrict__ ob,
                            float* __restrict__ out) {
  int pix = blockIdx.x;
  int c = threadIdx.x;
  int b = pix >> 12, rr = pix & (T_ - 1);
  int h = rr >> 6, wq = rr & 63;
  int t1 = wq * 64 + h;                 // inverse of column-major scan
  long base = (long)b * T_;
  float v = xf[(base + t1) * CK_ + c] +
            xf[(base + (T_ - 1 - t1)) * CK_ + (128 + c)];
  __shared__ float s0[128], s1[128];
  __shared__ float mu_s, rs_s;
  s0[c] = v; s1[c] = v * v; __syncthreads();
  for (int st = 64; st; st >>= 1) {
    if (c < st) { s0[c] += s0[c + st]; s1[c] += s1[c + st]; }
    __syncthreads();
  }
  if (!c) {
    float mu = s0[0] * (1.f / 128.f);
    float var = s1[0] * (1.f / 128.f) - mu * mu;
    mu_s = mu; rs_s = rsqrtf(var + EPS_);
  }
  __syncthreads();
  long oidx = ((long)(b * C_ + c)) * T_ + rr;
  out[oidx] = (v - mu_s) * rs_s * ow[c] + ob[c] + xin[oidx];
}

// ---------------------------------------------------------------------------
// Host orchestration
// ---------------------------------------------------------------------------
enum {
  IN_X = 0, IN_COND, IN_XLN_W, IN_XLN_B, IN_XCONV_W, IN_CA1_W, IN_CA1_B,
  IN_CA2_W, IN_CA2_B, IN_COND1_W, IN_COND1_B, IN_CONDLN_W, IN_CONDLN_B,
  IN_CONDDW_W, IN_TM_MAA_X, IN_TM_MAA_W, IN_TM_MAA_K, IN_TM_MAA_V,
  IN_TM_MAA_R, IN_TM_MAA_G, IN_TM_MAA_W1, IN_TM_MAA_W2, IN_TM_DECAY,
  IN_TM_DW1, IN_TM_DW2, IN_TM_FAAAA, IN_TM_WR, IN_TM_WK, IN_TM_WV,
  IN_TM_WG, IN_TM_WO, IN_TM_LNX_W, IN_TM_LNX_B, IN_CM_MAA_K, IN_CM_MAA_R,
  IN_CM_WK, IN_CM_WV, IN_CM_WR, IN_OUTLN_W, IN_OUTLN_B
};

extern "C" void kernel_launch(void* const* d_in, const int* in_sizes, int n_in,
                              void* d_out, int out_size, void* d_ws, size_t ws_size,
                              hipStream_t stream) {
  (void)in_sizes; (void)n_in; (void)out_size; (void)ws_size;
  const float* F[40];
  for (int i = 0; i < 40; ++i) F[i] = (const float*)d_in[i];

  // ---- workspace bump allocator -------------------------------------------
  char* ws = (char*)d_ws;
  size_t o = 0;
  auto alloc = [&](size_t bytes) -> char* {
    char* r = ws + o;
    o = (o + bytes + 255) & ~(size_t)255;
    return r;
  };
  const size_t F32M = (size_t)M_ * CK_ * 4;     // 33.5 MB
  const size_t F16M = (size_t)M_ * CK_ * 2;     // 16.8 MB
  const size_t IMG  = (size_t)B_ * C_ * T_ * 4; // 16.8 MB

  _Float16* w1t  = (_Float16*)alloc((size_t)192 * 256 * 2);
  _Float16* w2t  = (_Float16*)alloc((size_t)5 * 256 * 32 * 2);
  _Float16* wrt  = (_Float16*)alloc((size_t)256 * 256 * 2);
  _Float16* wkt  = (_Float16*)alloc((size_t)256 * 256 * 2);
  _Float16* wvt  = (_Float16*)alloc((size_t)256 * 256 * 2);
  _Float16* wgt  = (_Float16*)alloc((size_t)256 * 256 * 2);
  _Float16* wot  = (_Float16*)alloc((size_t)256 * 256 * 2);
  _Float16* dw1t = (_Float16*)alloc((size_t)64 * 256 * 2);
  _Float16* dw2t = (_Float16*)alloc((size_t)256 * 64 * 2);
  _Float16* cmkt = (_Float16*)alloc((size_t)512 * 256 * 2);
  _Float16* cmvt = (_Float16*)alloc((size_t)256 * 512 * 2);
  _Float16* cmrt = (_Float16*)alloc((size_t)256 * 256 * 2);
  float* statsX = (float*)alloc(B_ * 2 * 4);
  float* statsC = (float*)alloc(B_ * 2 * 4);
  float* xmax   = (float*)alloc(B_ * C_ * 4);
  float* gate   = (float*)alloc(B_ * C_ * 4);

  char* regA = alloc(IMG);    // xn   -> z16    -> yg16
  char* regB = alloc(IMG);    // cnd0 -> m16 (M x 192 f16)
  char* regC = alloc(IMG);    // s    -> h16 (M x 64 f16)
  char* regX = alloc(F32M);   // X32  -> y32
  char* mix[5]; for (int i = 0; i < 5; ++i) mix[i] = alloc(F16M); // xw,xk,xv,xr,xg
  char* rk[5];  for (int i = 0; i < 5; ++i) rk[i]  = alloc(F32M); // r,k,v,g,w

  float*    xn   = (float*)regA;
  float*    cnd0 = (float*)regB;
  float*    sbuf = (float*)regC;
  float*    X32  = (float*)regX;
  _Float16* z16  = (_Float16*)regA;
  _Float16* m16  = (_Float16*)regB;
  _Float16* h16  = (_Float16*)regC;
  float*    r32 = (float*)rk[0]; float* k32 = (float*)rk[1];
  float*    v32 = (float*)rk[2]; float* g32 = (float*)rk[3];
  float*    w32 = (float*)rk[4];
  float*    Sseg = (float*)mix[0];
  float*    Sini = (float*)mix[1];
  float*    Dseg = (float*)mix[2];
  float*    y32  = (float*)regX;
  _Float16* yg16 = (_Float16*)regA;
  float*    xattn = (float*)rk[0];
  _Float16* xk16c = (_Float16*)rk[1];
  _Float16* xr16c = (_Float16*)(rk[1] + F16M);
  _Float16* hk16  = (_Float16*)rk[2];
  float*    kv32  = (float*)rk[4];
  float*    xf32  = (float*)rk[3];

  auto prep = [&](const float* src, _Float16* dst, int K, int N, int Np) {
    long tot = (long)Np * K;
    wprep<<<(unsigned)((tot + 255) / 256), 256, 0, stream>>>(src, dst, K, N, Np);
  };
  auto gemm = [&](const _Float16* Ap, int lda, const _Float16* Bp,
                  void* Cp, int N, int K, int ldc, int mode,
                  const void* a0, const float* a1) {
    dim3 g(N / 64, M_ / 256);
    size_t sh = (size_t)128 * K;                 // 64 rows * K halfs * 2B
    gemm_wmma<<<g, 256, sh, stream>>>(Ap, lda, Bp, Cp, ldc, K, mode, a0, a1);
  };
  const unsigned EW = (unsigned)(((long)B_ * C_ * T_ + 255) / 256);
  const unsigned EM = (unsigned)(((long)M_ * CK_ + 255) / 256);

  // ---- 0) weight prep -----------------------------------------------------
  prep(F[IN_TM_MAA_W1], w1t, 256, 160, 192);
  for (int f = 0; f < 5; ++f)
    prep(F[IN_TM_MAA_W2] + (size_t)f * 32 * 256, w2t + (size_t)f * 256 * 32, 32, 256, 256);
  prep(F[IN_TM_WR], wrt, 256, 256, 256);
  prep(F[IN_TM_WK], wkt, 256, 256, 256);
  prep(F[IN_TM_WV], wvt, 256, 256, 256);
  prep(F[IN_TM_WG], wgt, 256, 256, 256);
  prep(F[IN_TM_WO], wot, 256, 256, 256);
  prep(F[IN_TM_DW1], dw1t, 256, 64, 64);
  prep(F[IN_TM_DW2], dw2t, 64, 256, 256);
  prep(F[IN_CM_WK], cmkt, 256, 512, 512);
  prep(F[IN_CM_WV], cmvt, 512, 256, 256);
  prep(F[IN_CM_WR], cmrt, 256, 256, 256);

  // ---- 1) spatial front end ----------------------------------------------
  gn_stats<<<B_, 256, 0, stream>>>(F[IN_X], (long)C_ * T_, statsX);
  xnorm_semax<<<B_ * C_, 256, 0, stream>>>(F[IN_X], statsX, F[IN_XLN_W], F[IN_XLN_B], xn, xmax);
  se_mlp<<<B_, 128, 0, stream>>>(xmax, F[IN_CA1_W], F[IN_CA1_B], F[IN_CA2_W], F[IN_CA2_B], gate);
  cond1x1<<<EW, 256, 0, stream>>>(F[IN_COND], F[IN_COND1_W], F[IN_COND1_B], cnd0);
  gn_stats<<<B_, 256, 0, stream>>>(cnd0, (long)C_ * T_, statsC);
  fuse_spatial<<<EW, 256, 0, stream>>>(xn, cnd0, statsC, F[IN_XCONV_W], F[IN_CONDDW_W],
                                       F[IN_CONDLN_W], F[IN_CONDLN_B], gate, sbuf);
  cross_scan<<<EM, 256, 0, stream>>>(sbuf, X32);

  // ---- 2) time-mix lora + projections (WMMA) ------------------------------
  mix16<<<EM, 256, 0, stream>>>(X32, F[IN_TM_MAA_X], z16);
  gemm(z16, 256, w1t, m16, 192, 256, 192, /*tanh f16*/2, nullptr, nullptr);
  const float* coefs[5] = { F[IN_TM_MAA_W], F[IN_TM_MAA_K], F[IN_TM_MAA_V],
                            F[IN_TM_MAA_R], F[IN_TM_MAA_G] };
  for (int f = 0; f < 5; ++f)
    gemm(m16 + f * 32, 192, w2t + (size_t)f * 256 * 32,
         mix[f], 256, 32, 256, /*lora-mix f16*/3, X32, coefs[f]);

  gemm((_Float16*)mix[3], 256, wrt, r32, 256, 256, 256, 0, nullptr, nullptr);
  gemm((_Float16*)mix[1], 256, wkt, k32, 256, 256, 256, 0, nullptr, nullptr);
  gemm((_Float16*)mix[2], 256, wvt, v32, 256, 256, 256, 0, nullptr, nullptr);
  gemm((_Float16*)mix[4], 256, wgt, g32, 256, 256, 256, /*silu*/1, nullptr, nullptr);
  gemm((_Float16*)mix[0], 256, dw1t, h16, 64, 256, 64, /*tanh f16*/2, nullptr, nullptr);
  gemm(h16, 64, dw2t, w32, 256, 64, 256, /*decay*/4, nullptr, F[IN_TM_DECAY]);

  // ---- 3) segmented WKV scan ----------------------------------------------
  wkv_pass1<<<dim3(NSEG, B_ * NH_), 32, 0, stream>>>(k32, v32, w32, Sseg, Dseg);
  wkv_pass2<<<B_ * NH_, 32, 0, stream>>>(Sseg, Dseg, Sini);
  wkv_pass3<<<dim3(NSEG, B_ * NH_), 32, 0, stream>>>(r32, k32, v32, w32,
                                                     F[IN_TM_FAAAA], Sini, y32);

  // ---- 4) head-LN * g, output projection ----------------------------------
  lnx_gate<<<M_, 256, 0, stream>>>(y32, g32, F[IN_TM_LNX_W], F[IN_TM_LNX_B], yg16);
  gemm(yg16, 256, wot, xattn, 256, 256, 256, 0, nullptr, nullptr);

  // ---- 5) channel-mix (WMMA) ----------------------------------------------
  mix16<<<EM, 256, 0, stream>>>(xattn, F[IN_CM_MAA_K], xk16c);
  mix16<<<EM, 256, 0, stream>>>(xattn, F[IN_CM_MAA_R], xr16c);
  gemm(xk16c, 256, cmkt, hk16, 512, 256, 512, /*relu^2 f16*/6, nullptr, nullptr);
  gemm(hk16, 512, cmvt, kv32, 256, 512, 256, 0, nullptr, nullptr);
  gemm(xr16c, 256, cmrt, xf32, 256, 256, 256, /*sig*aux*/7, kv32, nullptr);

  // ---- 6) cross-merge + out-LN + residual ---------------------------------
  merge_outln<<<B_ * T_, 128, 0, stream>>>(xf32, F[IN_X], F[IN_OUTLN_W],
                                           F[IN_OUTLN_B], (float*)d_out);
}
